// KPlaneField_10187662426200
// MI455X (gfx1250) — matrix operand order repaired and checked by reading, not compiled, not run on hardware
//
#include <hip/hip_runtime.h>
#include <hip/hip_bf16.h>
#include <math.h>

typedef __attribute__((ext_vector_type(16))) _Float16 v16h;
typedef __attribute__((ext_vector_type(8)))  _Float16 v8h;
typedef __attribute__((ext_vector_type(8)))  float    v8f;

struct GridPtrs { const float* g[24]; };

#define SPB 64      // samples per block
#define THREADS 256 // 8 waves (wave32)

__global__ __launch_bounds__(THREADS)
void kplane_fused(const float* __restrict__ pts,
                  const float* __restrict__ ts,
                  GridPtrs gp,
                  const float* __restrict__ W1,   // (128,64) row-major
                  const float* __restrict__ W2,   // (64,16)  row-major
                  float* __restrict__ out,        // (N,16)
                  int n_samp_per_ray)
{
    __shared__ alignas(32) _Float16 featsL[SPB * 128];  // 16 KB
    __shared__ alignas(32) _Float16 hL[SPB * 64];       //  8 KB
    __shared__ alignas(32) _Float16 W1T[64 * 128];      // 16 KB (W1T[n][k])
    __shared__ alignas(32) _Float16 W2T[16 * 64];       //  2 KB (W2T[n][k])

    const int tid    = threadIdx.x;
    const int lane   = tid & 31;
    const int wv     = tid >> 5;
    const int rowSel = lane >> 4;   // 0 for lanes 0-15, 1 for lanes 16-31
    const int lid16  = lane & 15;

    // ---- stage 0: weights -> LDS, transposed, f32 -> f16 ----
    #pragma unroll
    for (int i = 0; i < 32; ++i) {
        int idx = tid + i * THREADS;        // 8192 elements
        int n = idx >> 7;                   // 0..63
        int k = idx & 127;                  // 0..127
        W1T[n * 128 + k] = (_Float16)W1[k * 64 + n];
    }
    #pragma unroll
    for (int i = 0; i < 4; ++i) {
        int idx = tid + i * THREADS;        // 1024 elements
        int n = idx >> 6;                   // 0..15
        int k = idx & 63;                   // 0..63
        W2T[n * 64 + k] = (_Float16)W2[k * 16 + n];
    }

    // ---- stage 1: plane gather + bilinear, lane == channel ----
    // COMBS = (0,1)(0,2)(0,3)(1,2)(1,3)(2,3); W = reso[c0] (always 64*m), H = reso[c1]
    const int CB0[6] = {0, 0, 0, 1, 1, 2};
    const int CB1[6] = {1, 2, 3, 2, 3, 3};

    #pragma unroll 1
    for (int i = 0; i < SPB / 8; ++i) {
        const int sl = wv * 8 + i;
        const int S  = blockIdx.x * SPB + sl;
        float p4[4];
        p4[0] = pts[S * 3 + 0] * 2.0f - 1.0f;
        p4[1] = pts[S * 3 + 1] * 2.0f - 1.0f;
        p4[2] = pts[S * 3 + 2] * 2.0f - 1.0f;
        p4[3] = ts[S / n_samp_per_ray];     // already in [-1,1]

        #pragma unroll
        for (int sc = 0; sc < 4; ++sc) {
            const int m64 = 64 << sc;
            float acc = 1.0f;
            #pragma unroll
            for (int ci = 0; ci < 6; ++ci) {
                const int Wd = m64;
                const int Hd = (CB1[ci] == 3) ? 150 : m64;
                const float u = p4[CB0[ci]];
                const float v = p4[CB1[ci]];
                float x = (u + 1.0f) * 0.5f * (float)(Wd - 1);
                float y = (v + 1.0f) * 0.5f * (float)(Hd - 1);
                float xf = floorf(x), yf = floorf(y);
                float wx = x - xf,    wy = y - yf;
                int x0 = (int)xf; x0 = x0 < 0 ? 0 : (x0 > Wd - 1 ? Wd - 1 : x0);
                int y0 = (int)yf; y0 = y0 < 0 ? 0 : (y0 > Hd - 1 ? Hd - 1 : y0);
                int x1 = (x0 + 1 > Wd - 1) ? Wd - 1 : x0 + 1;
                int y1 = (y0 + 1 > Hd - 1) ? Hd - 1 : y0 + 1;
                const float* g = gp.g[sc * 6 + ci] + (size_t)lane * (size_t)(Hd * Wd);
                float g00 = g[y0 * Wd + x0];
                float g01 = g[y0 * Wd + x1];
                float g10 = g[y1 * Wd + x0];
                float g11 = g[y1 * Wd + x1];
                float sv = g00 * (1.0f - wx) * (1.0f - wy)
                         + g01 * wx          * (1.0f - wy)
                         + g10 * (1.0f - wx) * wy
                         + g11 * wx          * wy;
                acc *= sv;
            }
            featsL[sl * 128 + sc * 32 + lane] = (_Float16)acc;
        }
    }

    __syncthreads();

    // ---- stage 2: layer 1  feats(64x128) @ W1(128x64), relu, -> hL (f16) ----
    #pragma unroll
    for (int t = 0; t < 2; ++t) {
        const int tile = wv * 2 + t;        // 16 tiles over 8 waves
        const int Mt = tile >> 2, Nt = tile & 3;
        v8f acc = {};
        const _Float16* arow = &featsL[(Mt * 16 + lid16) * 128];
        const _Float16* brow = &W1T[(Nt * 16 + lid16) * 128];
        #pragma unroll
        for (int kk = 0; kk < 4; ++kk) {
            // A fragment: lane row = lane%16,
            // K = sel*16 + 0..15 contiguous (matches B)
            const int kbA = kk * 32 + rowSel * 16;
            v8h alo = *(const v8h*)(arow + kbA);
            v8h ahi = *(const v8h*)(arow + kbA + 8);
            v16h a = __builtin_shufflevector(alo, ahi,
                      0,1,2,3,4,5,6,7,8,9,10,11,12,13,14,15);
            // B fragment: lane col = lane%16, K = sel*16 + 0..15 contiguous
            const int kbB = kk * 32 + rowSel * 16;
            v16h b = *(const v16h*)(brow + kbB);
            acc = __builtin_amdgcn_wmma_f32_16x16x32_f16(
                      false, a, false, b, (short)0, acc, false, false);
        }
        #pragma unroll
        for (int r = 0; r < 8; ++r) {
            const int row = Mt * 16 + r + rowSel * 8;   // C/D layout
            const int col = Nt * 16 + lid16;
            float hv = acc[r];
            hL[row * 64 + col] = (_Float16)(hv > 0.0f ? hv : 0.0f);
        }
    }

    __syncthreads();

    // ---- stage 3: layer 2  h(64x64) @ W2(64x16), exp+reorder, store ----
    if (wv < 4) {
        const int Mt = wv;
        v8f acc = {};
        const _Float16* arow = &hL[(Mt * 16 + lid16) * 64];
        const _Float16* brow = &W2T[lid16 * 64];
        #pragma unroll
        for (int kk = 0; kk < 2; ++kk) {
            const int kbA = kk * 32 + rowSel * 16;
            v8h alo = *(const v8h*)(arow + kbA);
            v8h ahi = *(const v8h*)(arow + kbA + 8);
            v16h a = __builtin_shufflevector(alo, ahi,
                      0,1,2,3,4,5,6,7,8,9,10,11,12,13,14,15);
            const int kbB = kk * 32 + rowSel * 16;
            v16h b = *(const v16h*)(brow + kbB);
            acc = __builtin_amdgcn_wmma_f32_16x16x32_f16(
                      false, a, false, b, (short)0, acc, false, false);
        }
        const int col = lid16;
        #pragma unroll
        for (int r = 0; r < 8; ++r) {
            const int row = Mt * 16 + r + rowSel * 8;
            const long S = (long)blockIdx.x * SPB + row;
            float v = acc[r];
            if (col == 15) out[S * 16 + 0] = expf(v);   // density = exp(out[:,15])
            else           out[S * 16 + col + 1] = v;   // geo shifted by 1
        }
    }
}

extern "C" void kernel_launch(void* const* d_in, const int* in_sizes, int n_in,
                              void* d_out, int out_size, void* d_ws, size_t ws_size,
                              hipStream_t stream) {
    (void)d_ws; (void)ws_size; (void)n_in; (void)out_size;
    const float* pts = (const float*)d_in[0];
    const float* ts  = (const float*)d_in[1];
    GridPtrs gp;
    for (int i = 0; i < 24; ++i) gp.g[i] = (const float*)d_in[2 + i];
    const float* W1 = (const float*)d_in[26];
    const float* W2 = (const float*)d_in[27];
    float* out = (float*)d_out;

    const int total_samples = in_sizes[0] / 3;          // n_rays * n_samples
    const int n_rays        = in_sizes[1];
    const int nsamp         = total_samples / n_rays;   // samples per ray
    const int blocks        = total_samples / SPB;

    kplane_fused<<<blocks, THREADS, 0, stream>>>(pts, ts, gp, W1, W2, out, nsamp);
}
